// CausalSelfAttention_65987877535899
// MI455X (gfx1250) — matrix operand order repaired and checked by reading, not compiled, not run on hardware
//
#include <hip/hip_runtime.h>

// ---------------------------------------------------------------------------
// Causal self-attention (GPT-2 style) for MI455X / gfx1250, wave32 + WMMA +
// Tensor Data Mover (TDM) tile staging.
// B=2, T=4096, C=768, NH=12, HD=64.
// Pipeline:
//   1) f32 -> f16 conversions (x plain; weights converted AND transposed so
//      GEMM B-tiles are contiguous 2D tiles -> TDM-stageable, and the LDS
//      layout is directly the WMMA B-operand (N-major) layout)
//   2) qkv = x @ w_attn + b_attn   (WMMA GEMM, TDM double-buffered staging)
//   3) flash attention per head    (WMMA QK^T / P@V, online softmax,
//                                   K tile staged via TDM)
//   4) out = y @ w_proj + b_proj   (WMMA GEMM, f32 out)
// ---------------------------------------------------------------------------

typedef __attribute__((ext_vector_type(16))) _Float16 v16h;
typedef __attribute__((ext_vector_type(8)))  _Float16 v8h;
typedef __attribute__((ext_vector_type(8)))  float    v8f;
typedef __attribute__((ext_vector_type(4)))  unsigned int v4u;
typedef __attribute__((ext_vector_type(8)))  int v8i;
typedef __attribute__((ext_vector_type(4)))  int v4i;

#ifndef __has_builtin
#define __has_builtin(x) 0
#endif
#if defined(__HIP_DEVICE_COMPILE__) && \
    __has_builtin(__builtin_amdgcn_tensor_load_to_lds) && \
    __has_builtin(__builtin_amdgcn_s_wait_tensorcnt)
#define USE_TDM 1
#else
#define USE_TDM 0
#endif

#define WMMA_F32_F16(a, b, c) \
  __builtin_amdgcn_wmma_f32_16x16x32_f16(false, (a), false, (b), (short)0, (c), false, false)

// ---------------------------------------------------------------------------
// TDM: DMA a 2D f16 tile [tile_d1 rows][tile_d0 elems] from global (row
// stride `stride_e` elements) into LDS (contiguous row-major). Issued once
// per wave (EXEC ignored); caller gates to one wave and waits TENSORcnt.
// D# packing per CDNA5 ISA 8.3/8.4 (2D: groups 2/3 zero, tile_dim2=0 unused).
// This toolchain exposes the 6-arg builtin (extra int32x8 group, zero-filled).
// ---------------------------------------------------------------------------
#if USE_TDM
__device__ __forceinline__ void tdm_load_tile_f16(void* lds_dst, const void* gsrc,
                                                  unsigned tile_d0, unsigned tile_d1,
                                                  unsigned stride_e) {
  const unsigned lds_off = (unsigned)(size_t)lds_dst;  // low 32b of flat = LDS offset
  const unsigned long long ga = (unsigned long long)(size_t)gsrc;

  v4u g0;
  g0[0] = 1u;                                            // count=1 (valid), user mode
  g0[1] = lds_off;                                       // lds_addr [63:32]
  g0[2] = (unsigned)(ga & 0xffffffffull);                // global_addr[31:0]
  g0[3] = (unsigned)((ga >> 32) & 0x1ffffffull)          // global_addr[56:32]
        | (2u << 30);                                    // type=2 ("image")

  unsigned d[8];
#pragma unroll
  for (int i = 0; i < 8; ++i) d[i] = 0u;
  d[0] = (1u << 16);                                     // data_size=1 -> 2 bytes
  // tensor_dim0 [79:48] = stride_e (row length; tiles never OOB here)
  d[1] = (stride_e & 0xffffu) << 16;
  d[2] = ((stride_e >> 16) & 0xffffu)
       | ((tile_d1 & 0xffffu) << 16);                    // tensor_dim1[15:0] = rows
  d[3] = ((tile_d1 >> 16) & 0xffffu)
       | ((tile_d0 & 0xffffu) << 16);                    // tile_dim0 [127:112]
  d[4] = (tile_d1 & 0xffffu);                            // tile_dim1; tile_dim2=0
  d[5] = stride_e;                                       // tensor_dim0_stride[31:0]
  d[6] = 0u;                                             // stride[47:32]; dim1_stride lo
  d[7] = 0u;

  v8i g1;
#pragma unroll
  for (int i = 0; i < 8; ++i) g1[i] = (int)d[i];
  const v4i z4 = {0, 0, 0, 0};
  const v8i z8 = {0, 0, 0, 0, 0, 0, 0, 0};
  __builtin_amdgcn_tensor_load_to_lds(g0, g1, z4, z4, z8, 0);
}
#endif

// A-operand fragment (16x32 f16), row-major source with `stride` elements/row.
// Layout used here (not ISA table order): lane<16 -> row=lane, K = kb+0..15 contiguous; lane>=16 -> K+16.
__device__ __forceinline__ v16h frag_a_ld(const _Float16* __restrict__ base,
                                          int row0, int stride, int kb, int lane) {
  const int half = lane >> 4, lr = lane & 15;
  const _Float16* p = base + (size_t)(row0 + lr) * stride + kb + half * 16;
  v8h lo = *(const v8h*)p;          // K = kb + half*16 + 0..7
  v8h hi = *(const v8h*)(p + 8);    // K = kb + half*16 + 8..15
  v16h r;
#pragma unroll
  for (int i = 0; i < 8; ++i) { r[i] = lo[i]; r[i + 8] = hi[i]; }
  return r;
}

// B-operand fragment (32x16 f16) from an N-major buffer: (k,n) at base[n*stride+k].
// Layout used here (not ISA table order): lane<16 -> col=lane, K = kb+0..15 contiguous; lane>=16 -> K+16.
__device__ __forceinline__ v16h frag_b_ld(const _Float16* __restrict__ base,
                                          int col0, int stride, int kb, int lane) {
  const int half = lane >> 4, lr = lane & 15;
  const _Float16* p = base + (size_t)(col0 + lr) * stride + kb + half * 16;
  v8h lo = *(const v8h*)p;
  v8h hi = *(const v8h*)(p + 8);
  v16h r;
#pragma unroll
  for (int i = 0; i < 8; ++i) { r[i] = lo[i]; r[i + 8] = hi[i]; }
  return r;
}

// ---------------------------------------------------------------------------
// elementwise f32 -> f16
// ---------------------------------------------------------------------------
__global__ void f32_to_f16_kernel(const float* __restrict__ in,
                                  _Float16* __restrict__ out, int n) {
  int i = blockIdx.x * blockDim.x + threadIdx.x;
  if (i < n) out[i] = (_Float16)in[i];
}

// f32 [R][Ncols] -> f16 transposed [Ncols][R]
__global__ void f32_to_f16_transpose_kernel(const float* __restrict__ in,
                                            _Float16* __restrict__ out,
                                            int R, int Ncols) {
  int i = blockIdx.x * blockDim.x + threadIdx.x;
  if (i < R * Ncols) {
    int r = i / Ncols, c = i % Ncols;
    out[(size_t)c * R + r] = (_Float16)in[i];
  }
}

// ---------------------------------------------------------------------------
// Tiled WMMA GEMM: C[M,N] = A[M,K] * Bt[N,K]^T + bias[N]
// A row-major f16 (ld=K); Bt row-major N-major f16 (ld=K) == B transposed.
// Block: 256 thr = 8 waves (2x4), tile 64(M) x 128(N), K stepped by 32,
// double-buffered LDS staged by TDM (wave 0 issues, TENSORcnt + barrier).
// Requires M%64==0, N%128==0, K%32==0.
// ---------------------------------------------------------------------------
template <bool OUT_F16>
__global__ __launch_bounds__(256) void gemm_wmma_kernel(
    const _Float16* __restrict__ A, const _Float16* __restrict__ Bt,
    const float* __restrict__ bias, void* __restrict__ Cout,
    int M, int N, int K) {
  __shared__ _Float16 As[2][64 * 32];    // A tile, row-major [64][32]
  __shared__ _Float16 Bts[2][128 * 32];  // B tile, N-major [128][32]

  const int tid = threadIdx.x;
  const int lane = tid & 31, wave = tid >> 5;
  const int wr = wave >> 2, wc = wave & 3;  // 2 x 4 wave grid
  const int mB = blockIdx.y * 64, nB = blockIdx.x * 128;
  const int half = lane >> 4, lr = lane & 15;

  const v8f vzero = {0.f, 0.f, 0.f, 0.f, 0.f, 0.f, 0.f, 0.f};
  v8f acc[2][2];
#pragma unroll
  for (int ri = 0; ri < 2; ++ri)
#pragma unroll
    for (int ci = 0; ci < 2; ++ci) acc[ri][ci] = vzero;

  const int nkt = K >> 5;

#if USE_TDM
  if (wave == 0) {
    tdm_load_tile_f16(&As[0][0],  A  + (size_t)mB * K, 32, 64,  (unsigned)K);
    tdm_load_tile_f16(&Bts[0][0], Bt + (size_t)nB * K, 32, 128, (unsigned)K);
  }
#endif

  for (int t = 0; t < nkt; ++t) {
    const int buf = t & 1;
#if USE_TDM
    if (wave == 0) __builtin_amdgcn_s_wait_tensorcnt((short)0);
    __syncthreads();  // buf[t&1] ready for all; prior reads of buf[(t+1)&1] done
    if (wave == 0 && t + 1 < nkt) {
      const int kn = (t + 1) * 32;
      tdm_load_tile_f16(&As[(t + 1) & 1][0],  A  + (size_t)mB * K + kn, 32, 64,  (unsigned)K);
      tdm_load_tile_f16(&Bts[(t + 1) & 1][0], Bt + (size_t)nB * K + kn, 32, 128, (unsigned)K);
    }
#else
    __syncthreads();
    {  // manual A stage: 64x32, each thread 8 contiguous halves
      int row = tid >> 2, cb = (tid & 3) * 8;
      v8h v = *(const v8h*)(A + (size_t)(mB + row) * K + t * 32 + cb);
      *(v8h*)(&As[buf][row * 32 + cb]) = v;
    }
    {  // manual Bt stage: 128x32, each thread 16 contiguous halves
      int n = tid >> 1, cb = (tid & 1) * 16;
      const _Float16* src = Bt + (size_t)(nB + n) * K + t * 32 + cb;
      *(v8h*)(&Bts[buf][n * 32 + cb])     = *(const v8h*)(src);
      *(v8h*)(&Bts[buf][n * 32 + cb + 8]) = *(const v8h*)(src + 8);
    }
    __syncthreads();
#endif

    v16h a0 = frag_a_ld(&As[buf][0], wr * 32 + 0,  32, 0, lane);
    v16h a1 = frag_a_ld(&As[buf][0], wr * 32 + 16, 32, 0, lane);
    v16h b0 = frag_b_ld(&Bts[buf][0], wc * 32 + 0,  32, 0, lane);
    v16h b1 = frag_b_ld(&Bts[buf][0], wc * 32 + 16, 32, 0, lane);

    acc[0][0] = WMMA_F32_F16(a0, b0, acc[0][0]);
    acc[0][1] = WMMA_F32_F16(a0, b1, acc[0][1]);
    acc[1][0] = WMMA_F32_F16(a1, b0, acc[1][0]);
    acc[1][1] = WMMA_F32_F16(a1, b1, acc[1][1]);
  }

  // Epilogue. D layout: comp i, lane<16 -> row i, lane>=16 -> row 8+i; col=lane&15.
#pragma unroll
  for (int ri = 0; ri < 2; ++ri) {
#pragma unroll
    for (int ci = 0; ci < 2; ++ci) {
      const int n = nB + wc * 32 + ci * 16 + lr;
      const float bv = bias[n];
#pragma unroll
      for (int i = 0; i < 8; ++i) {
        const int m = mB + wr * 32 + ri * 16 + half * 8 + i;
        const float v = acc[ri][ci][i] + bv;
        if (OUT_F16)
          ((_Float16*)Cout)[(size_t)m * N + n] = (_Float16)v;
        else
          ((float*)Cout)[(size_t)m * N + n] = v;
      }
    }
  }
}

// ---------------------------------------------------------------------------
// Flash attention, causal. Grid: x = T/128 q-blocks, y = B*NH heads.
// 256 threads = 8 waves; wave w handles q rows [qblock*128 + w*16, +16).
// KV tiles of 32 keys; K tile staged via TDM (row-major 32x64), V tile
// transposed manually (TDM cannot transpose). qkv layout: [B, T, 3C].
// Output y: f16 [B, T, C].
// ---------------------------------------------------------------------------
__global__ __launch_bounds__(256) void attn_flash_kernel(
    const _Float16* __restrict__ qkv, _Float16* __restrict__ y) {
  constexpr int T = 4096, C3 = 2304, Cc = 768, NH = 12, HD = 64;

  __shared__ _Float16 Ks[32 * 64];      // K tile   [tk=32][d=64] (row-major)
  __shared__ _Float16 Vts[64 * 32];     // V tile^T [d=64][tk=32]
  __shared__ _Float16 Ps[8 * 16 * 32];  // per-wave P staging [16][32]

  const int tid = threadIdx.x, lane = tid & 31, wave = tid >> 5;
  const int half = lane >> 4, lr = lane & 15;
  const int qblock = blockIdx.x;
  const int bh = blockIdx.y;
  const int b = bh / NH, h = bh % NH;

  const size_t headBase = (size_t)b * T * C3 + h * HD;
  const _Float16* Qp = qkv + headBase;
  const _Float16* Kp = qkv + headBase + Cc;
  const _Float16* Vp = qkv + headBase + 2 * Cc;

  const int q0 = qblock * 128 + wave * 16;

  // Q fragments for this wave's 16 rows (HD=64 -> two k=32 steps), kept in regs.
  const v16h qa0 = frag_a_ld(Qp, q0, C3, 0, lane);
  const v16h qa1 = frag_a_ld(Qp, q0, C3, 32, lane);

  const v8f vzero = {0.f, 0.f, 0.f, 0.f, 0.f, 0.f, 0.f, 0.f};
  v8f o[4];
#pragma unroll
  for (int j = 0; j < 4; ++j) o[j] = vzero;

  float mrun[8], lrun[8];
#pragma unroll
  for (int i = 0; i < 8; ++i) { mrun[i] = -__builtin_inff(); lrun[i] = 0.f; }

  _Float16* Pw = Ps + wave * (16 * 32);
  const int ntile = qblock * 4 + 4;  // uniform over the block (causal bound)
  const float scale = 0.125f;        // 1/sqrt(64)

  for (int t = 0; t < ntile; ++t) {
    const int kv = t * 32;
    __syncthreads();  // previous iteration's reads of Ks/Vts complete
#if USE_TDM
    if (wave == 0)
      tdm_load_tile_f16(&Ks[0], Kp + (size_t)kv * C3, 64, 32, (unsigned)C3);
#else
    {  // stage K tile: 32 rows x 64 d
      int r = tid >> 3, db = (tid & 7) * 8;
      v8h v = *(const v8h*)(Kp + (size_t)(kv + r) * C3 + db);
      *(v8h*)(Ks + r * 64 + db) = v;
    }
#endif
    {  // stage V tile transposed: Vts[d][tk]
      int r = tid >> 3, db = (tid & 7) * 8;
      v8h v = *(const v8h*)(Vp + (size_t)(kv + r) * C3 + db);
#pragma unroll
      for (int i = 0; i < 8; ++i) Vts[(db + i) * 32 + r] = v[i];
    }
#if USE_TDM
    if (wave == 0) __builtin_amdgcn_s_wait_tensorcnt((short)0);
#endif
    __syncthreads();

    // S = Q K^T : two 16x16 output tiles (key cols 0-15 and 16-31)
    v8f s0 = vzero, s1 = vzero;
    {
      v16h kb00 = frag_b_ld(Ks, 0,  HD, 0,  lane);
      v16h kb01 = frag_b_ld(Ks, 0,  HD, 32, lane);
      s0 = WMMA_F32_F16(qa0, kb00, s0);
      s0 = WMMA_F32_F16(qa1, kb01, s0);
      v16h kb10 = frag_b_ld(Ks, 16, HD, 0,  lane);
      v16h kb11 = frag_b_ld(Ks, 16, HD, 32, lane);
      s1 = WMMA_F32_F16(qa0, kb10, s1);
      s1 = WMMA_F32_F16(qa1, kb11, s1);
    }

    // scale + causal mask; rows: q0 + half*8 + i, cols: kv+lr and kv+16+lr
    const int c0 = kv + lr, c1 = c0 + 16;
    float mx[8];
#pragma unroll
    for (int i = 0; i < 8; ++i) {
      const int row = q0 + half * 8 + i;
      float x0 = (c0 <= row) ? s0[i] * scale : -__builtin_inff();
      float x1 = (c1 <= row) ? s1[i] * scale : -__builtin_inff();
      s0[i] = x0; s1[i] = x1;
      mx[i] = fmaxf(x0, x1);
    }
    // row max across the 16 lanes of each half-wave group
#pragma unroll
    for (int d = 8; d >= 1; d >>= 1)
#pragma unroll
      for (int i = 0; i < 8; ++i)
        mx[i] = fmaxf(mx[i], __shfl_xor(mx[i], d, 32));

    float rs[8], alpha[8];
#pragma unroll
    for (int i = 0; i < 8; ++i) {
      const float mn = fmaxf(mrun[i], mx[i]);
      const float p0 = __expf(s0[i] - mn);
      const float p1 = __expf(s1[i] - mn);
      alpha[i] = __expf(mrun[i] - mn);
      mrun[i] = mn;
      s0[i] = p0; s1[i] = p1;
      rs[i] = p0 + p1;
    }
#pragma unroll
    for (int d = 8; d >= 1; d >>= 1)
#pragma unroll
      for (int i = 0; i < 8; ++i)
        rs[i] += __shfl_xor(rs[i], d, 32);
#pragma unroll
    for (int i = 0; i < 8; ++i) lrun[i] = lrun[i] * alpha[i] + rs[i];
#pragma unroll
    for (int j = 0; j < 4; ++j)
#pragma unroll
      for (int i = 0; i < 8; ++i) o[j][i] *= alpha[i];

    // Reshape P (D layout) -> A layout via LDS staging, f16
#pragma unroll
    for (int i = 0; i < 8; ++i) {
      const int row = half * 8 + i;
      Pw[row * 32 + lr]      = (_Float16)s0[i];
      Pw[row * 32 + 16 + lr] = (_Float16)s1[i];
    }
    __syncthreads();

    const v16h pa = frag_a_ld(Pw, 0, 32, 0, lane);
#pragma unroll
    for (int j = 0; j < 4; ++j) {
      v16h vb = frag_b_ld(Vts, j * 16, 32, 0, lane);
      o[j] = WMMA_F32_F16(pa, vb, o[j]);
    }
  }

  // Normalize and write y (f16) at [b, row, h*64 + d]
#pragma unroll
  for (int j = 0; j < 4; ++j) {
#pragma unroll
    for (int i = 0; i < 8; ++i) {
      const int row = q0 + half * 8 + i;
      const int col = h * HD + j * 16 + lr;
      const float v = o[j][i] / lrun[i];
      y[((size_t)b * T + row) * Cc + col] = (_Float16)v;
    }
  }
}

// ---------------------------------------------------------------------------
// Host launch
// ---------------------------------------------------------------------------
extern "C" void kernel_launch(void* const* d_in, const int* in_sizes, int n_in,
                              void* d_out, int out_size, void* d_ws, size_t ws_size,
                              hipStream_t stream) {
  constexpr int Bb = 2, Tt = 4096, Cc = 768, NH = 12;
  const float* x      = (const float*)d_in[0];  // [B,T,C]
  const float* w_attn = (const float*)d_in[1];  // [C,3C]
  const float* b_attn = (const float*)d_in[2];  // [3C]
  const float* w_proj = (const float*)d_in[3];  // [C,C]
  const float* b_proj = (const float*)d_in[4];  // [C]
  float* out = (float*)d_out;                   // [B,T,C]

  const size_t nX   = (size_t)Bb * Tt * Cc;       // 6,291,456
  const size_t nWa  = (size_t)Cc * 3 * Cc;        // 1,769,472
  const size_t nWp  = (size_t)Cc * Cc;            //   589,824
  const size_t nQKV = (size_t)Bb * Tt * 3 * Cc;   // 18,874,368

  _Float16* xh   = (_Float16*)d_ws;               // [B*T, C]
  _Float16* wat  = xh + nX;                       // w_attn^T [3C, C]
  _Float16* wpt  = wat + nWa;                     // w_proj^T [C, C]
  _Float16* qkvh = wpt + nWp;                     // [B*T, 3C]
  _Float16* yh   = qkvh + nQKV;                   // [B*T, C]

  // 1) conversions (weights also transposed: Bt[N][K])
  f32_to_f16_kernel<<<(unsigned)((nX + 255) / 256), 256, 0, stream>>>(x, xh, (int)nX);
  f32_to_f16_transpose_kernel<<<(unsigned)((nWa + 255) / 256), 256, 0, stream>>>(
      w_attn, wat, Cc, 3 * Cc);
  f32_to_f16_transpose_kernel<<<(unsigned)((nWp + 255) / 256), 256, 0, stream>>>(
      w_proj, wpt, Cc, Cc);

  // 2) qkv = x @ w_attn + b_attn   (M=8192, N=2304, K=768), f16 out
  {
    dim3 grid(3 * Cc / 128, (Bb * Tt) / 64);  // (18, 128)
    gemm_wmma_kernel<true><<<grid, 256, 0, stream>>>(
        xh, wat, b_attn, (void*)qkvh, Bb * Tt, 3 * Cc, Cc);
  }

  // 3) flash attention -> yh (f16 [B,T,C])
  {
    dim3 grid(Tt / 128, Bb * NH);  // (32, 24)
    attn_flash_kernel<<<grid, 256, 0, stream>>>(qkvh, yh);
  }

  // 4) out = y @ w_proj + b_proj  (M=8192, N=768, K=768), f32 out
  {
    dim3 grid(Cc / 128, (Bb * Tt) / 64);  // (6, 128)
    gemm_wmma_kernel<false><<<grid, 256, 0, stream>>>(
        yh, wpt, b_proj, (void*)out, Bb * Tt, Cc, Cc);
  }
}
